// ScalarSoftmaxQuantization_14104672600377
// MI455X (gfx1250) — matrix operand order repaired and checked by reading, not compiled, not run on hardware
//
#include <hip/hip_runtime.h>

typedef __attribute__((ext_vector_type(16))) _Float16 v16h;
typedef __attribute__((ext_vector_type(8)))  float    v8f;

#define ALPHA_C  (-10.0f)
#define LOG2E_C  (1.4426950408889634f)
#define EPS_C    (1e-10f)

// ds_swizzle group-of-32 xor pattern: imm = (xor_mask<<10) | and_mask(0x1F)
template <int IMM>
__device__ __forceinline__ float swz(float v) {
  return __builtin_bit_cast(
      float, __builtin_amdgcn_ds_swizzle(__builtin_bit_cast(int, v), IMM));
}
#define SWZ_X1  0x041F
#define SWZ_X2  0x081F
#define SWZ_X4  0x101F
#define SWZ_X8  0x201F
#define SWZ_X16 0x401F

// ---------------------------------------------------------------------------
// Main kernel: each wave handles a 32-element pair per iteration as two
// interleaved 16-element WMMA tiles (tile0 = even elements, tile1 = odd).
// exp/sqrt on VALU, the K-contraction (numerator + denominator) on WMMA/XDL.
// ---------------------------------------------------------------------------
__global__ __launch_bounds__(256) void ssq_main(const float* __restrict__ x,
                                                const float* __restrict__ bins,
                                                float* __restrict__ out,
                                                float* __restrict__ partials,
                                                int nPairs) {
  __shared__ float nsLds[8][2][16][2];  // per-wave num/S broadcast (2 KB)
  __shared__ float red[2][8][32];       // per-block k-reduction (2 KB)

  const int tid  = threadIdx.x;
  const int lane = tid & 31;
  const int wv   = tid >> 5;
  const int hi   = (lane >> 4) & 1;     // which half of the wave
  const int n16  = lane & 15;           // element-in-tile / N index

  // bins are wave-uniform -> SGPRs via s_load
  float sb[32];
#pragma unroll
  for (int k = 0; k < 32; ++k) sb[k] = bins[k];
  float sumB = 0.f;
#pragma unroll
  for (int k = 0; k < 32; ++k) sumB += sb[k];
  const float epsB = EPS_C * sumB;

  // Per-lane k mapping matching the 16-bit A-matrix VGPR layout (ISA 7.12.2):
  //  lo lanes: j=0..7 -> k=j,   j=8..15 -> k=j+8   ({0-7,16-23})
  //  hi lanes: j=0..7 -> k=j+8, j=8..15 -> k=j+16  ({8-15,24-31})
  float bk[16];
#pragma unroll
  for (int j = 0; j < 16; ++j) {
    const int klo = j + ((j < 8) ? 0 : 8);
    bk[j] = hi ? sb[klo + 8] : sb[klo];
  }

  // Constant B operand for V_WMMA_F32_16X16X32_F16:
  //  B[k][0] = bins[k], B[k][1] = 1.0, other columns 0.
  v16h Bm;
#pragma unroll
  for (int j = 0; j < 16; ++j) {
    float v = 0.f;
    if (n16 == 0)      v = sb[j + ((j < 8) ? 0 : 8) + hi * 8];
    else if (n16 == 1) v = 1.0f;
    Bm[j] = (_Float16)v;
  }

  float accP[16], accS[16];
#pragma unroll
  for (int j = 0; j < 16; ++j) { accP[j] = 0.f; accS[j] = 0.f; }

  const int waveGlobal = blockIdx.x * 8 + wv;
  const int waveCount  = gridDim.x * 8;

  for (int pair = waveGlobal; pair < nPairs; pair += waveCount) {
    const int base = pair << 5;                       // 32 elements per pair
    const float2 xv = *(const float2*)(x + base + 2 * n16);  // b64, coalesced

    // ---- tile 0 (even elements) ----
    float dmin0 = 1e30f, dmin1 = 1e30f;
    float d0[16], d1[16];
#pragma unroll
    for (int j = 0; j < 16; ++j) {
      d0[j] = __builtin_fabsf(xv.x - bk[j]);
      d1[j] = __builtin_fabsf(xv.y - bk[j]);
      dmin0 = fminf(dmin0, d0[j]);
      dmin1 = fminf(dmin1, d1[j]);
    }
    dmin0 = fminf(dmin0, swz<SWZ_X16>(dmin0));  // combine the two k-halves
    dmin1 = fminf(dmin1, swz<SWZ_X16>(dmin1));

    float e0[16], e1[16];
    v16h Am0, Am1;
#pragma unroll
    for (int j = 0; j < 16; ++j) {
      e0[j] = __builtin_amdgcn_exp2f((ALPHA_C * LOG2E_C) * (d0[j] - dmin0));
      e1[j] = __builtin_amdgcn_exp2f((ALPHA_C * LOG2E_C) * (d1[j] - dmin1));
      Am0[j] = (_Float16)e0[j];
      Am1[j] = (_Float16)e1[j];
    }

    // C[m][0] = num_m = sum_k e'*bins_k ; C[m][1] = S_m = sum_k e'
    v8f C0 = {}, C1 = {};
    C0 = __builtin_amdgcn_wmma_f32_16x16x32_f16(false, Am0, false, Bm,
                                                (short)0, C0, false, false);
    C1 = __builtin_amdgcn_wmma_f32_16x16x32_f16(false, Am1, false, Bm,
                                                (short)0, C1, false, false);

    // num/S rows live in lanes 0,16 (N=0) and 1,17 (N=1): scatter to LDS,
    // wait for the DS pipe, gather per element.
    if (n16 < 2) {
#pragma unroll
      for (int r = 0; r < 8; ++r) {
        nsLds[wv][0][hi * 8 + r][n16] = C0[r];
        nsLds[wv][1][hi * 8 + r][n16] = C1[r];
      }
    }
    __builtin_amdgcn_wave_barrier();
    asm volatile("s_wait_dscnt 0" ::: "memory");
    const float2 ns0 = *(const float2*)&nsLds[wv][0][n16][0];
    const float2 ns1 = *(const float2*)&nsLds[wv][1][n16][0];
    __builtin_amdgcn_wave_barrier();

    const float rS0 = __builtin_amdgcn_rcpf(ns0.y);
    const float rS1 = __builtin_amdgcn_rcpf(ns1.y);

    // bit_code = (sum_k soft*b_k) + EPS*sum(b_k); lo half stores b64
    if (!hi) {
      float2 bc;
      bc.x = ns0.x * rS0 + epsB;
      bc.y = ns1.x * rS1 + epsB;
      *(float2*)(out + base + 2 * n16) = bc;
    }

    // per-k running sums of assignment and sqrt(assignment)
#pragma unroll
    for (int j = 0; j < 16; ++j) {
      const float a0 = e0[j] * rS0 + EPS_C;
      const float a1 = e1[j] * rS1 + EPS_C;
      accP[j] += a0 + a1;
      accS[j] += __builtin_amdgcn_sqrtf(a0) + __builtin_amdgcn_sqrtf(a1);
    }
  }

  // --- teardown: half-wave swizzle tree (lanes in a half share one k-set) ---
#pragma unroll
  for (int j = 0; j < 16; ++j) {
    accP[j] += swz<SWZ_X1>(accP[j]);
    accP[j] += swz<SWZ_X2>(accP[j]);
    accP[j] += swz<SWZ_X4>(accP[j]);
    accP[j] += swz<SWZ_X8>(accP[j]);
    accS[j] += swz<SWZ_X1>(accS[j]);
    accS[j] += swz<SWZ_X2>(accS[j]);
    accS[j] += swz<SWZ_X4>(accS[j]);
    accS[j] += swz<SWZ_X8>(accS[j]);
  }
  if (n16 == 0) {                 // lanes 0 and 16 hold the half-wave sums
#pragma unroll
    for (int j = 0; j < 16; ++j) {
      const int k = j + ((j < 8) ? 0 : 8) + hi * 8;
      red[0][wv][k] = accP[j];
      red[1][wv][k] = accS[j];
    }
  }
  __syncthreads();
  if (tid < 64) {                 // deterministic fixed-order block partials
    const int which = tid >> 5, k = tid & 31;
    float s = 0.f;
#pragma unroll
    for (int w = 0; w < 8; ++w) s += red[which][w][k];
    partials[blockIdx.x * 64 + tid] = s;
  }
}

// ---------------------------------------------------------------------------
// Finalize: deterministic reduction over block partials; entropy + quant loss.
// ---------------------------------------------------------------------------
__global__ __launch_bounds__(64) void ssq_final(const float* __restrict__ partials,
                                                float* __restrict__ out,
                                                int nBlocks, int nElems) {
  __shared__ float acc[64];
  const int t = threadIdx.x;
  float s = 0.f;
  for (int b = 0; b < nBlocks; ++b) s += partials[b * 64 + t];
  acc[t] = s;
  __syncthreads();
  if (t == 0) {
    const float invN = 1.0f / (float)nElems;
    float ent = 0.f, quant = 0.f;
    for (int k = 0; k < 32; ++k) {
      const float p = acc[k] * invN;
      ent   -= p * __logf(p);
      quant += acc[32 + k] * invN;
    }
    out[nElems + 0] = ent;    // weighted_code_entropy[0,0]
    out[nElems + 1] = 0.0f;   // TAU
    out[nElems + 2] = quant;  // weighted_quant_loss[0,0]
    out[nElems + 3] = 0.0f;   // TAU2
  }
}

extern "C" void kernel_launch(void* const* d_in, const int* in_sizes, int n_in,
                              void* d_out, int out_size, void* d_ws, size_t ws_size,
                              hipStream_t stream) {
  const float* x    = (const float*)d_in[0];
  const float* bins = (const float*)d_in[1];
  float* out = (float*)d_out;
  float* ws  = (float*)d_ws;

  const int nElems = in_sizes[0];        // B*F*L (multiple of 32 here)
  const int nPairs = nElems >> 5;        // 32 elements (two WMMA tiles) per pair

  int grid = 1024;                       // 8192 waves -> 8 pairs each at N=2M
  const int maxByWs = (int)(ws_size / (64 * sizeof(float)));
  if (grid > maxByWs && maxByWs > 0) grid = maxByWs;
  const int maxByWork = (nPairs + 7) / 8;
  if (grid > maxByWork) grid = maxByWork;
  if (grid < 1) grid = 1;

  ssq_main<<<grid, 256, 0, stream>>>(x, bins, out, ws, nPairs);
  ssq_final<<<1, 64, 0, stream>>>(ws, out, grid, nElems);
}
